// SINQLinear_2241972929044
// MI455X (gfx1250) — matrix operand order repaired and checked by reading, not compiled, not run on hardware
//
#include <hip/hip_runtime.h>

typedef __attribute__((ext_vector_type(16))) _Float16 v16h;
typedef __attribute__((ext_vector_type(8)))  _Float16 v8h;
typedef __attribute__((ext_vector_type(8)))  float    v8f;

#define IN_F          4096
#define OUT_F         4096
#define GROUPS_PER_ROW (IN_F / 64)

#define BM 128
#define BN 128
#define BK 64
#define LDS_STRIDE 72   // halfs; 144 B row stride (multiple of 16 B, bank-skewed)

// D = A(x, f16) * B(dequant W, f16)^T + bias, f32 accumulate via v_wmma_f32_16x16x32_f16.
__global__ __launch_bounds__(256)
void sinq_wmma_gemm(const float* __restrict__ x,
                    const int*   __restrict__ Wq,     // one byte per int32 element
                    const float* __restrict__ scale,  // [262144]
                    const float* __restrict__ zero,   // [262144]
                    const float* __restrict__ bias,   // [4096]
                    float* __restrict__ out)          // [8192, 4096]
{
    __shared__ _Float16 As[BM * LDS_STRIDE];
    __shared__ _Float16 Bs[BN * LDS_STRIDE];

    const int tid  = threadIdx.x;
    const int lane = tid & 31;
    const int wave = tid >> 5;
    const int wm   = wave >> 2;      // 0..1 : 64-row slab
    const int wn   = wave & 3;       // 0..3 : 32-col slab

    const int m0 = blockIdx.y * BM;  // row tile into flattened [8192, 4096] x
    const int n0 = blockIdx.x * BN;  // out-feature tile

    // ---- cooperative loader mapping: 2 threads per row, 32 K-elements each
    const int lr  = tid >> 1;            // 0..127
    const int lc0 = (tid & 1) * 32;      // 0 or 32

    const int  o     = n0 + lr;                       // out feature dequantized by this thread
    const long wqRow = (long)(o & 2047) * IN_F;       // byte-row in Wq
    const int  shift = (o < 2048) ? 4 : 0;            // hi nibbles for o<2048, else lo

    // ---- per-lane WMMA fragment base addresses (in halfs)
    int aRow[4], bCol[2];
#pragma unroll
    for (int mi = 0; mi < 4; ++mi)
        aRow[mi] = (wm * 64 + mi * 16 + (lane & 15)) * LDS_STRIDE;
#pragma unroll
    for (int ni = 0; ni < 2; ++ni)
        bCol[ni] = (wn * 32 + ni * 16 + (lane & 15)) * LDS_STRIDE;
    const int aKsel = (lane >> 4) * 8;    // A: hi half-wave starts at K=8
    const int bKsel = (lane >> 4) * 8;    // B: hi half-wave starts at K=8

    v8f acc[4][2];
#pragma unroll
    for (int mi = 0; mi < 4; ++mi)
#pragma unroll
        for (int ni = 0; ni < 2; ++ni)
#pragma unroll
            for (int v = 0; v < 8; ++v)
                acc[mi][ni][v] = 0.0f;

    for (int kb = 0; kb < IN_F; kb += BK) {
        // ================= global -> registers =================
        const float4* xsrc = (const float4*)(x + (long)(m0 + lr) * IN_F + kb + lc0);
        float4 af[8];
#pragma unroll
        for (int j = 0; j < 8; ++j) af[j] = xsrc[j];

        const int4* wsrc = (const int4*)(Wq + wqRow + kb + lc0);
        int4 bq[8];
#pragma unroll
        for (int j = 0; j < 8; ++j) bq[j] = wsrc[j];

        const int   g = o * GROUPS_PER_ROW + (kb >> 6);
        const float s = scale[g];
        const float z = zero[g];

        // prefetch next K tile (speculative; dropped if OOB)
        __builtin_prefetch(xsrc + 8, 0, 0);
        __builtin_prefetch(wsrc + 8, 0, 0);

        __syncthreads();   // previous iteration's LDS reads complete

        // ================= registers -> LDS (f16) =================
        _Float16* adst = &As[lr * LDS_STRIDE + lc0];
#pragma unroll
        for (int j = 0; j < 4; ++j) {
            v8h hv;
            hv[0] = (_Float16)af[2*j].x;   hv[1] = (_Float16)af[2*j].y;
            hv[2] = (_Float16)af[2*j].z;   hv[3] = (_Float16)af[2*j].w;
            hv[4] = (_Float16)af[2*j+1].x; hv[5] = (_Float16)af[2*j+1].y;
            hv[6] = (_Float16)af[2*j+1].z; hv[7] = (_Float16)af[2*j+1].w;
            *(v8h*)&adst[8 * j] = hv;
        }

        _Float16* bdst = &Bs[lr * LDS_STRIDE + lc0];
#pragma unroll
        for (int j = 0; j < 4; ++j) {
            const int q[8] = { bq[2*j].x,   bq[2*j].y,   bq[2*j].z,   bq[2*j].w,
                               bq[2*j+1].x, bq[2*j+1].y, bq[2*j+1].z, bq[2*j+1].w };
            v8h hv;
#pragma unroll
            for (int e = 0; e < 8; ++e)
                hv[e] = (_Float16)(((float)((q[e] >> shift) & 0xF) - z) * s);
            *(v8h*)&bdst[8 * j] = hv;
        }

        __syncthreads();   // tile staged

        // ================= LDS -> frags -> WMMA =================
#pragma unroll
        for (int ks = 0; ks < 2; ++ks) {
            const int ka  = ks * 32 + aKsel;
            const int kbb = ks * 32 + bKsel;

            v16h a[4], b[2];
#pragma unroll
            for (int mi = 0; mi < 4; ++mi) {
                union { v16h f; v8h h[2]; } u;
                u.h[0] = *(const v8h*)&As[aRow[mi] + ka];        // K+0..7
                u.h[1] = *(const v8h*)&As[aRow[mi] + ka + 16];   // K+16..23
                a[mi] = u.f;
            }
#pragma unroll
            for (int ni = 0; ni < 2; ++ni) {
                union { v16h f; v8h h[2]; } u;
                u.h[0] = *(const v8h*)&Bs[bCol[ni] + kbb];       // K+0..7
                u.h[1] = *(const v8h*)&Bs[bCol[ni] + kbb + 16];  // K+16..23
                b[ni] = u.f;
            }

#pragma unroll
            for (int mi = 0; mi < 4; ++mi)
#pragma unroll
                for (int ni = 0; ni < 2; ++ni)
                    acc[mi][ni] = __builtin_amdgcn_wmma_f32_16x16x32_f16(
                        false, a[mi], false, b[ni],
                        (short)0, acc[mi][ni], false, false);
        }
    }

    // ================= epilogue: C layout -> global, fused bias =================
#pragma unroll
    for (int ni = 0; ni < 2; ++ni) {
        const int   col = n0 + wn * 32 + ni * 16 + (lane & 15);
        const float bv  = bias[col];
#pragma unroll
        for (int mi = 0; mi < 4; ++mi) {
            const int rbase = m0 + wm * 64 + mi * 16 + (lane >> 4) * 8;
#pragma unroll
            for (int v = 0; v < 8; ++v)
                out[(long)(rbase + v) * OUT_F + col] = acc[mi][ni][v] + bv;
        }
    }
}

extern "C" void kernel_launch(void* const* d_in, const int* in_sizes, int n_in,
                              void* d_out, int out_size, void* d_ws, size_t ws_size,
                              hipStream_t stream) {
    (void)in_sizes; (void)n_in; (void)out_size; (void)d_ws; (void)ws_size;
    const float* x     = (const float*)d_in[0];
    const int*   Wq    = (const int*)  d_in[1];
    const float* scale = (const float*)d_in[2];
    const float* zero  = (const float*)d_in[3];
    const float* bias  = (const float*)d_in[4];
    float*       out   = (float*)d_out;

    dim3 grid(OUT_F / BN, (4 * 2048) / BM);   // 32 x 64 tiles
    dim3 block(256);
    sinq_wmma_gemm<<<grid, block, 0, stream>>>(x, Wq, scale, zero, bias, out);
}
